// ExecPolicyNetwork_7232724927129
// MI455X (gfx1250) — matrix operand order repaired and checked by reading, not compiled, not run on hardware
//
#include <hip/hip_runtime.h>
#include <hip/hip_bf16.h>
#include <math.h>

typedef __attribute__((ext_vector_type(16))) _Float16 v16h;
typedef __attribute__((ext_vector_type(8)))  float    v8f;

#define COMP_THREADS 256
#define COMP_ITEMS   16
#define COMP_CHUNK   (COMP_THREADS * COMP_ITEMS)

// Branch-free tanh: 1 - 2*rcp(exp(2x)+1). Native v_exp_f32 + v_rcp_f32,
// correct limits at +/-inf, propagates NaN, no EXEC divergence, no IEEE div.
__device__ __forceinline__ float fast_tanh(float x) {
  float e = __expf(2.0f * x);
  return fmaf(-2.0f, __builtin_amdgcn_rcpf(e + 1.0f), 1.0f);
}

// ---------------- Stream compaction (ordered nonzero) ----------------

__global__ void k_block_count(const unsigned char* __restrict__ mask,
                              const int* __restrict__ jidp,
                              int E, int* __restrict__ blockSums) {
  const int jid = *jidp;
  const unsigned char* row = mask + (size_t)jid * (size_t)E;
  int t = threadIdx.x;
  int base = blockIdx.x * COMP_CHUNK + t * COMP_ITEMS;
  int cnt = 0;
  for (int i = 0; i < COMP_ITEMS; ++i) {
    int e = base + i;
    if (e < E && row[e]) ++cnt;
  }
  __shared__ int red[COMP_THREADS];
  red[t] = cnt;
  __syncthreads();
  for (int off = COMP_THREADS >> 1; off > 0; off >>= 1) {
    if (t < off) red[t] += red[t + off];
    __syncthreads();
  }
  if (t == 0) blockSums[blockIdx.x] = red[0];
}

__global__ void k_scan_sums(const int* __restrict__ blockSums,
                            int* __restrict__ blockOffs, int n) {
  if (threadIdx.x == 0 && blockIdx.x == 0) {
    int acc = 0;
    for (int b = 0; b < n; ++b) { blockOffs[b] = acc; acc += blockSums[b]; }
  }
}

__global__ void k_scatter_acts(const unsigned char* __restrict__ mask,
                               const int* __restrict__ jidp,
                               int E, const int* __restrict__ blockOffs,
                               float* acts, int K) {
  const int jid = *jidp;
  const unsigned char* row = mask + (size_t)jid * (size_t)E;
  int t = threadIdx.x;
  int base = blockIdx.x * COMP_CHUNK + t * COMP_ITEMS;
  int cnt = 0;
  for (int i = 0; i < COMP_ITEMS; ++i) {
    int e = base + i;
    if (e < E && row[e]) ++cnt;
  }
  __shared__ int scan[COMP_THREADS];
  scan[t] = cnt;
  __syncthreads();
  for (int off = 1; off < COMP_THREADS; off <<= 1) {
    int v = (t >= off) ? scan[t - off] : 0;
    __syncthreads();
    scan[t] += v;
    __syncthreads();
  }
  int rank = blockOffs[blockIdx.x] + scan[t] - cnt;
  float fE = (float)E;
  for (int i = 0; i < COMP_ITEMS; ++i) {
    int e = base + i;
    if (e < E && row[e]) {
      if (rank < K) acts[rank] = (float)e / fE;
      ++rank;
    }
  }
}

// ---------------- Layer-1 constant fold: c1 = W1[:, :IN-1] @ base + b1 ----------------

__global__ void k_prep(const float* __restrict__ x, const int* __restrict__ ptr,
                       const float* __restrict__ hdag, const float* __restrict__ hglob,
                       const float* __restrict__ W1, const float* __restrict__ b1,
                       const int* __restrict__ jidp,
                       int J, int NDF, int DEM, int GEM, int IN, int xElems,
                       float* __restrict__ c1, float* __restrict__ wl) {
  int n = threadIdx.x;                 // blockDim.x == H1 (128)
  int jid = *jidp;
  int rows = ptr[J];                   // total node rows
  int F = xElems / rows;               // node feature stride
  int xRow = ptr[jid];
  const float* w = W1 + (size_t)n * (size_t)IN;
  float acc = b1[n];
  for (int i = 0; i < IN - 1; ++i) {
    float bv;
    if (i < NDF)            bv = x[(size_t)xRow * F + i];
    else if (i < NDF + DEM) bv = hdag[(size_t)jid * DEM + (i - NDF)];
    else                    bv = hglob[i - NDF - DEM];
    acc = fmaf(bv, w[i], acc);
  }
  c1[n] = acc;
  wl[n] = w[IN - 1];
}

// ---------------- Fused MLP: h1=tanh(c1+a*wl); h2=tanh(h1@W2^T+b2) via WMMA; score=h2@W3^T+b3
// Assumes H1=128, H2=64 (matches reference). One 16-row tile per wave.

__global__ void __launch_bounds__(256)
k_mlp_wmma(float* io,                               // in: acts[K]; out: scores[K]
           const float* __restrict__ c1g, const float* __restrict__ wlg,
           const float* __restrict__ W2, const float* __restrict__ b2,
           const float* __restrict__ W3, const float* __restrict__ b3,
           int K) {
  __shared__ __align__(32) _Float16 sW2[64 * 128];  // [n][k], row-major like W2
  __shared__ float sc1[128], swl[128], sb2[64], sw3[64];
  int t = threadIdx.x;
  for (int i = t; i < 64 * 128; i += 256) sW2[i] = (_Float16)W2[i];
  if (t < 128) { sc1[t] = c1g[t]; swl[t] = wlg[t]; }
  if (t < 64)  { sb2[t] = b2[t];  sw3[t] = W3[t]; }
  __syncthreads();

  const float b3v = b3[0];
  int lane = t & 31;
  int wave = t >> 5;
  int tile = blockIdx.x * 8 + wave;
  int numTiles = (K + 15) >> 4;
  if (tile >= numTiles) return;       // uniform per wave

  int ln = lane & 15;
  int hs = lane >> 4;                 // lane half: 0 or 1
  int tileBase = tile << 4;

  int aRow = tileBase + ln;           // A-matrix row for this lane (both halves)
  float a = io[aRow < K ? aRow : (K - 1)];

  v8f acc[4] = {};                    // 4 N-tiles of 16 columns

  int c0 = hs ? 16 : 0;               // A: this code's contiguous K run {c0..c0+15} per lane half, same order as B
#pragma unroll
  for (int kt = 0; kt < 4; ++kt) {
    int kb = kt * 32 + c0;
    v16h A;
#pragma unroll
    for (int j = 0; j < 8; ++j) {
      int k = kb + j;
      A[j] = (_Float16)fast_tanh(fmaf(a, swl[k], sc1[k]));
    }
#pragma unroll
    for (int j = 0; j < 8; ++j) {
      int k = kb + 8 + j;
      A[8 + j] = (_Float16)fast_tanh(fmaf(a, swl[k], sc1[k]));
    }
    int bOff = kt * 32 + (hs ? 16 : 0);   // B layout: contiguous K run per lane half
#pragma unroll
    for (int u = 0; u < 4; ++u) {
      int n = u * 16 + ln;                // B column = output neuron
      v16h B = *(const v16h*)&sW2[n * 128 + bOff];
      acc[u] = __builtin_amdgcn_wmma_f32_16x16x32_f16(
          false, A, false, B, (short)0, acc[u], false, false);
    }
  }

  // Layer-2 activation + layer-3 dot. D layout: VGPR r -> row r (lanes<16) / r+8, col = u*16+ln.
  float s[8];
#pragma unroll
  for (int r = 0; r < 8; ++r) s[r] = 0.0f;
#pragma unroll
  for (int u = 0; u < 4; ++u) {
    int n = u * 16 + ln;
    float bb = sb2[n], ww = sw3[n];
#pragma unroll
    for (int r = 0; r < 8; ++r)
      s[r] += fast_tanh(acc[u][r] + bb) * ww;
  }
#pragma unroll
  for (int r = 0; r < 8; ++r)
    for (int m = 1; m < 16; m <<= 1)
      s[r] += __shfl_xor(s[r], m, 32);

  if (ln == 0) {
    int rb = tileBase + (hs ? 8 : 0);
#pragma unroll
    for (int r = 0; r < 8; ++r)
      if (rb + r < K) io[rb + r] = s[r] + b3v;
  }
}

// ---------------- Host launcher ----------------

extern "C" void kernel_launch(void* const* d_in, const int* in_sizes, int n_in,
                              void* d_out, int out_size, void* d_ws, size_t ws_size,
                              hipStream_t stream) {
  const float* x        = (const float*)d_in[0];
  const int*   ptr      = (const int*)d_in[1];
  const unsigned char* mask = (const unsigned char*)d_in[2];  // jnp bool = 1 byte
  const float* hdag     = (const float*)d_in[3];
  const float* hglob    = (const float*)d_in[4];
  const float* W1       = (const float*)d_in[5];
  const float* b1       = (const float*)d_in[6];
  const float* W2       = (const float*)d_in[7];
  const float* b2       = (const float*)d_in[8];
  const float* W3       = (const float*)d_in[9];
  const float* b3       = (const float*)d_in[10];
  const int*   jidp     = (const int*)d_in[11];
  // d_in[12] = num_exec_acts (device scalar); host-side K == out_size

  int J   = in_sizes[1] - 1;            // 16
  int E   = in_sizes[2] / J;            // 200000
  int DEM = in_sizes[3] / J;            // 128
  int GEM = in_sizes[4];                // 128
  int H1  = in_sizes[6];                // 128
  int IN  = in_sizes[5] / H1;           // 260
  int NDF = IN - 1 - DEM - GEM;         // 3
  int K   = out_size;                   // 120000

  float* c1 = (float*)d_ws;
  float* wl = c1 + H1;
  int* blockSums = (int*)(wl + H1);
  int nBlk = (E + COMP_CHUNK - 1) / COMP_CHUNK;
  int* blockOffs = blockSums + nBlk;

  float* io = (float*)d_out;            // holds acts, then overwritten with scores

  k_block_count<<<nBlk, COMP_THREADS, 0, stream>>>(mask, jidp, E, blockSums);
  k_scan_sums<<<1, 1, 0, stream>>>(blockSums, blockOffs, nBlk);
  k_scatter_acts<<<nBlk, COMP_THREADS, 0, stream>>>(mask, jidp, E, blockOffs, io, K);
  k_prep<<<1, H1, 0, stream>>>(x, ptr, hdag, hglob, W1, b1, jidp,
                               J, NDF, DEM, GEM, IN, in_sizes[0], c1, wl);

  int numTiles = (K + 15) / 16;
  int mBlocks  = (numTiles + 7) / 8;    // 8 waves/block, 1 tile/wave
  k_mlp_wmma<<<mBlocks, 256, 0, stream>>>(io, c1, wl, W2, b2, W3, b3, K);
}
